// Global_Context_Module_20186346291782
// MI455X (gfx1250) — matrix operand order repaired and checked by reading, not compiled, not run on hardware
//
#include <hip/hip_runtime.h>
#include <math.h>

typedef __attribute__((ext_vector_type(16))) _Float16 v16h;
typedef __attribute__((ext_vector_type(4)))  _Float16 v4h;
typedef __attribute__((ext_vector_type(8)))  float    v8f;
typedef __attribute__((ext_vector_type(4)))  float    v4f;

#define LDT 40  // LDS tile pitch in halves (80B rows: 16B-aligned)

// A fragment for V_WMMA_F32_16X16X32_F16 (16x32 f16, one row per lane).
// Lanes 0-15: K=[0..7],[16..23]; lanes 16-31: K=[8..15],[24..31]. (ISA 7.12.2)
__device__ __forceinline__ v16h frag_a(const _Float16* base, int lane) {
  const int m  = lane & 15;
  const int kb = (lane < 16) ? 0 : 8;
  const _Float16* p = base + m * LDT + kb;
  v16h r;
#pragma unroll
  for (int e = 0; e < 8; ++e) r[e] = p[e];
#pragma unroll
  for (int e = 0; e < 8; ++e) r[8 + e] = p[16 + e];
  return r;
}

// B fragment (32x16 f16). Bs stored transposed: Bs[n][k] -> contiguous K per lane
// (two ds_load_b128). Lanes 0-15: K=[0..7],[16..23]; lanes 16-31: K=[8..15],[24..31].
__device__ __forceinline__ v16h frag_b(const _Float16* base, int lane) {
  const int n  = lane & 15;
  const int kb = (lane < 16) ? 0 : 8;
  const _Float16* p = base + n * LDT + kb;
  v16h r;
#pragma unroll
  for (int e = 0; e < 16; ++e) r[e] = p[(e < 8) ? e : (e + 8)];
  return r;
}

__device__ __forceinline__ v4h cvt4(v4f t) {
  v4h h;
  h[0] = (_Float16)t[0]; h[1] = (_Float16)t[1];
  h[2] = (_Float16)t[2]; h[3] = (_Float16)t[3];
  return h;
}

// K-contiguous staging: four 16B global loads (independent, latency-hidden),
// four 8B LDS stores of converted halves.
__device__ __forceinline__ void stage_k(_Float16* __restrict__ S, const float* gp,
                                        long gInc, int lds) {
  v4f t0 = *(const v4f*)(gp);
  v4f t1 = *(const v4f*)(gp + gInc);
  v4f t2 = *(const v4f*)(gp + 2 * gInc);
  v4f t3 = *(const v4f*)(gp + 3 * gInc);
  *(v4h*)(S + lds)            = cvt4(t0);
  *(v4h*)(S + lds + 16 * LDT) = cvt4(t1);
  *(v4h*)(S + lds + 32 * LDT) = cvt4(t2);
  *(v4h*)(S + lds + 48 * LDT) = cvt4(t3);
}

// M/N-contiguous staging (other stride is 1): four 16B loads along m/n, halves
// scattered down 4 LDS rows each.
__device__ __forceinline__ void stage_m(_Float16* __restrict__ S, const float* gp,
                                        long gInc, int lds) {
  v4f t0 = *(const v4f*)(gp);
  v4f t1 = *(const v4f*)(gp + gInc);
  v4f t2 = *(const v4f*)(gp + 2 * gInc);
  v4f t3 = *(const v4f*)(gp + 3 * gInc);
  _Float16* p = S + lds;
#pragma unroll
  for (int e = 0; e < 4; ++e) p[e * LDT] = (_Float16)t0[e];
#pragma unroll
  for (int e = 0; e < 4; ++e) p[8 + e * LDT] = (_Float16)t1[e];
#pragma unroll
  for (int e = 0; e < 4; ++e) p[16 + e * LDT] = (_Float16)t2[e];
#pragma unroll
  for (int e = 0; e < 4; ++e) p[24 + e * LDT] = (_Float16)t3[e];
}

// Generic strided batched GEMM: C[z][m][n] = alpha * (sum_k A[z][m][k]*B[z][k][n] + bias)
// 64x64 block tile, 4 waves, 2x2 register blocking per wave (4 WMMAs/K-step),
// double-buffered LDS, one barrier per K-step. AKF/BKF: operand's K-stride == 1
// (else its m/n stride is 1 -- true for every GEMM in this pipeline).
// biasMode: 0=none, 1=bias[n], 2=bias[m]. M,N multiples of 64; K multiple of 32.
template <bool AKF, bool BKF>
__global__ __launch_bounds__(128)
void gemm_wmma_f16(const float* __restrict__ A, const float* __restrict__ Bm,
                   float* __restrict__ Cm, const float* __restrict__ bias,
                   int biasMode, float alpha, int K,
                   long sAm, long sAk, long sBk, long sBn, long sCm, long sCn,
                   int zdiv, long sAb, long sAh, long sBb, long sBh, long sCb, long sCh)
{
  __shared__ __align__(16) _Float16 As[2][64 * LDT];  // As[m][k]
  __shared__ __align__(16) _Float16 Bs[2][64 * LDT];  // Bs[n][k] (transposed)

  const int tid  = threadIdx.x;
  const int lane = tid & 31;
  const int wid  = tid >> 5;
  const int tm = wid >> 1, tn = wid & 1;
  const int m0 = blockIdx.y * 64;
  const int n0 = blockIdx.x * 64;
  const int z  = blockIdx.z;
  const int zq = z / zdiv, zr = z % zdiv;
  const long offA = (long)zq * sAb + (long)zr * sAh;
  const long offB = (long)zq * sBb + (long)zr * sBh;
  const long offC = (long)zq * sCb + (long)zr * sCh;

  // Per-thread staging descriptors (quad-granular; computed once).
  const float* pa; long aInc; int aLds;
  if (AKF) {  // quads along k: r = tid>>3 (+16/j), kq = (tid&7)*4
    const int r = tid >> 3, kq = (tid & 7) * 4;
    pa = A + offA + (long)(m0 + r) * sAm + kq;
    aInc = 16 * sAm; aLds = r * LDT + kq;
  } else {    // sAm == 1; quads along m: r = 4*(tid&15), c = tid>>4 (+8/j)
    const int rq = tid & 15, c = tid >> 4;
    pa = A + offA + (long)(m0 + 4 * rq) + (long)c * sAk;
    aInc = 8 * sAk; aLds = (4 * rq) * LDT + c;
  }
  const long aK = 32 * sAk;

  const float* pb; long bInc; int bLds;
  if (BKF) {
    const int n = tid >> 3, kq = (tid & 7) * 4;
    pb = Bm + offB + (long)(n0 + n) * sBn + kq;
    bInc = 16 * sBn; bLds = n * LDT + kq;
  } else {    // sBn == 1
    const int nq = tid & 15, c = tid >> 4;
    pb = Bm + offB + (long)(n0 + 4 * nq) + (long)c * sBk;
    bInc = 8 * sBk; bLds = (4 * nq) * LDT + c;
  }
  const long bK = 32 * sBk;

  v8f acc[2][2] = {};

  const int nK = K >> 5;
  if (AKF) stage_k(As[0], pa, aInc, aLds); else stage_m(As[0], pa, aInc, aLds);
  if (BKF) stage_k(Bs[0], pb, bInc, bLds); else stage_m(Bs[0], pb, bInc, bLds);
  pa += aK; pb += bK;
  int cur = 0;
  for (int t = 0; t < nK; ++t) {
    __syncthreads();  // staging of buffer `cur` complete; prior reads of cur^1 done
    // Fragment loads first so the WMMA DS reads are not queued behind staging stores.
    v16h a0 = frag_a(&As[cur][(32 * tm) * LDT], lane);
    v16h a1 = frag_a(&As[cur][(32 * tm + 16) * LDT], lane);
    v16h b0 = frag_b(&Bs[cur][(32 * tn) * LDT], lane);
    v16h b1 = frag_b(&Bs[cur][(32 * tn + 16) * LDT], lane);
    if (t + 1 < nK) {
      if (AKF) stage_k(As[cur ^ 1], pa, aInc, aLds); else stage_m(As[cur ^ 1], pa, aInc, aLds);
      if (BKF) stage_k(Bs[cur ^ 1], pb, bInc, bLds); else stage_m(Bs[cur ^ 1], pb, bInc, bLds);
      pa += aK; pb += bK;
    }
    acc[0][0] = __builtin_amdgcn_wmma_f32_16x16x32_f16(false, a0, false, b0,
                                                       (short)0, acc[0][0], false, false);
    acc[0][1] = __builtin_amdgcn_wmma_f32_16x16x32_f16(false, a0, false, b1,
                                                       (short)0, acc[0][1], false, false);
    acc[1][0] = __builtin_amdgcn_wmma_f32_16x16x32_f16(false, a1, false, b0,
                                                       (short)0, acc[1][0], false, false);
    acc[1][1] = __builtin_amdgcn_wmma_f32_16x16x32_f16(false, a1, false, b1,
                                                       (short)0, acc[1][1], false, false);
    cur ^= 1;
  }

  // C/D layout: VGPR r holds M = r (lanes 0-15) / r+8 (lanes 16-31), N = lane&15.
  const int col = lane & 15;
  const int rhi = (lane >= 16) ? 8 : 0;
#pragma unroll
  for (int ti = 0; ti < 2; ++ti) {
#pragma unroll
    for (int tj = 0; tj < 2; ++tj) {
      const int gmb = m0 + 32 * tm + 16 * ti + rhi;
      const int gn  = n0 + 32 * tn + 16 * tj + col;
      const float bcol = (biasMode == 1) ? bias[gn] : 0.0f;
#pragma unroll
      for (int r = 0; r < 8; ++r) {
        const int gm = gmb + r;
        float v = acc[ti][tj][r];
        if (biasMode == 2) v += bias[gm];
        else               v += bcol;
        v *= alpha;
        Cm[offC + (long)gm * sCm + (long)gn * sCn] = v;
      }
    }
  }
}

__global__ void softmax_rows(float* __restrict__ data, int S) {
  float* row = data + (long)blockIdx.x * S;
  __shared__ float red[256];
  const int tid = threadIdx.x;
  float m = -INFINITY;
  for (int j = tid; j < S; j += 256) m = fmaxf(m, row[j]);
  red[tid] = m; __syncthreads();
  for (int s = 128; s > 0; s >>= 1) {
    if (tid < s) red[tid] = fmaxf(red[tid], red[tid + s]);
    __syncthreads();
  }
  m = red[0]; __syncthreads();
  float sum = 0.f;
  for (int j = tid; j < S; j += 256) {
    float e = __expf(row[j] - m);
    row[j] = e;
    sum += e;
  }
  red[tid] = sum; __syncthreads();
  for (int s = 128; s > 0; s >>= 1) {
    if (tid < s) red[tid] += red[tid + s];
    __syncthreads();
  }
  const float inv = 1.0f / red[0];
  for (int j = tid; j < S; j += 256) row[j] *= inv;
}

__global__ void copy_batched(const float* __restrict__ src, float* __restrict__ dst,
                             long nPerB, long sSrc, long sDst, int nb) {
  long i = (long)blockIdx.x * blockDim.x + threadIdx.x;
  if (i >= nPerB * nb) return;
  long b = i / nPerB;
  long j = i - b * nPerB;
  dst[b * sDst + j] = src[b * sSrc + j];
}

static inline void gemmL(hipStream_t st, const float* A, const float* B, float* C,
                         const float* bias, int biasMode, float alpha,
                         int M, int N, int K,
                         long sAm, long sAk, long sBk, long sBn, long sCm, long sCn,
                         int Z, int zdiv, long sAb, long sAh, long sBb, long sBh,
                         long sCb, long sCh) {
  dim3 g((unsigned)(N / 64), (unsigned)(M / 64), (unsigned)Z), blk(128);
  const bool akf = (sAk == 1), bkf = (sBk == 1);
  if (akf && bkf)
    gemm_wmma_f16<true, true><<<g, blk, 0, st>>>(A, B, C, bias, biasMode, alpha, K,
        sAm, sAk, sBk, sBn, sCm, sCn, zdiv, sAb, sAh, sBb, sBh, sCb, sCh);
  else if (akf)
    gemm_wmma_f16<true, false><<<g, blk, 0, st>>>(A, B, C, bias, biasMode, alpha, K,
        sAm, sAk, sBk, sBn, sCm, sCn, zdiv, sAb, sAh, sBb, sBh, sCb, sCh);
  else if (bkf)
    gemm_wmma_f16<false, true><<<g, blk, 0, st>>>(A, B, C, bias, biasMode, alpha, K,
        sAm, sAk, sBk, sBn, sCm, sCn, zdiv, sAb, sAh, sBb, sBh, sCb, sCh);
  else
    gemm_wmma_f16<false, false><<<g, blk, 0, st>>>(A, B, C, bias, biasMode, alpha, K,
        sAm, sAk, sBk, sBn, sCm, sCn, zdiv, sAb, sAh, sBb, sBh, sCb, sCh);
}

extern "C" void kernel_launch(void* const* d_in, const int* in_sizes, int n_in,
                              void* d_out, int out_size, void* d_ws, size_t ws_size,
                              hipStream_t stream) {
  (void)in_sizes; (void)n_in; (void)out_size;
  const int Bn = 2, NHh = 2;
  const int CHl[4] = {64, 128, 256, 512};
  const int HWl[4] = {128, 64, 32, 16};

  float* ws = (float*)d_ws;
  const long QMAX  = 1048576;  // max B*L*C floats (level 1)
  const long CATMX = 3145728;  // max B*3C*L floats (level 1)
  float* Qb   = ws;
  float* Kb   = ws + QMAX;
  float* Vb   = ws + 2 * QMAX;
  float* Ob   = ws + 3 * QMAX;
  float* catb = ws + 4 * QMAX;
  float* scb  = catb + CATMX;
  // scores region: whatever scratch remains beyond the fixed buffers (>= 8MB assumed)
  long scAvail = (long)(ws_size / sizeof(float)) - (4 * QMAX + CATMX);
  if (scAvail < 2097152) scAvail = 2097152;  // assume at least ~38MB total ws

  // level 0: pass-through
  hipMemcpyAsync(d_out, d_in[0], (size_t)Bn * 64 * 128 * 128 * sizeof(float),
                 hipMemcpyDeviceToDevice, stream);

  float* outP = (float*)d_out;
  long outOff = (long)Bn * 64 * 128 * 128;

  for (int lvl = 1; lvl <= 3; ++lvl) {
    const int  C  = CHl[lvl];
    const int  sp = HWl[lvl];
    const long L  = (long)sp * sp, S = L;
    const int  dh = C / NHh;
    const float scale = 1.0f / sqrtf((float)dh);
    const float* f1 = (const float*)d_in[lvl];
    const float* f2 = (const float*)d_in[4 + (lvl - 1) * 2];
    const float* f3 = (const float*)d_in[4 + (lvl - 1) * 2 + 1];
    const int wb = 10 + (lvl - 1) * 10;
    const float* sqw = (const float*)d_in[wb + 8];
    const float* sqb = (const float*)d_in[wb + 9];

    // cat[:, 0:C, :] = f1
    {
      long total = (long)Bn * C * L;
      copy_batched<<<dim3((unsigned)((total + 255) / 256)), dim3(256), 0, stream>>>(
          f1, catb, (long)C * L, (long)C * L, (long)3 * C * L, Bn);
    }

    for (int at = 0; at < 2; ++at) {
      const float* kv = at ? f3 : f2;
      const float* w  = (const float*)d_in[wb + at * 4 + 0];  // [3C, C]
      const float* bi = (const float*)d_in[wb + at * 4 + 1];  // [3C]
      const float* ow = (const float*)d_in[wb + at * 4 + 2];  // [C, C]
      const float* ob = (const float*)d_in[wb + at * 4 + 3];  // [C]
      const long c0 = (long)(at + 1) * C * L;  // att slot in cat

      // Q[b,l,c] = sum_c' f1[b,c',l]*wq[c,c'] + bq[c]
      gemmL(stream, f1, w, Qb, bi, 1, 1.0f, (int)L, C, C,
            1, L, 1, C, (long)C, 1,
            Bn, 1, (long)C * L, 0, 0, 0, (long)L * C, 0);
      // K
      gemmL(stream, kv, w + (long)C * C, Kb, bi + C, 1, 1.0f, (int)S, C, C,
            1, S, 1, C, (long)C, 1,
            Bn, 1, (long)C * S, 0, 0, 0, (long)S * C, 0);
      // V
      gemmL(stream, kv, w + (long)2 * C * C, Vb, bi + 2 * C, 1, 1.0f, (int)S, C, C,
            1, S, 1, C, (long)C, 1,
            Bn, 1, (long)C * S, 0, 0, 0, (long)S * C, 0);

      // attention, chunked over L so scores fit in the scratch that remains
      long R = scAvail / ((long)Bn * NHh * S);
      if (R > L) R = L;
      R &= ~63L;
      if (R < 64) R = 64;
      for (long l0 = 0; l0 < L; l0 += R) {
        // scores[z=(b,h)][m][s] = scale * Q[b,l0+m,h*dh+:] . K[b,s,h*dh+:]
        gemmL(stream, Qb + l0 * C, Kb, scb, nullptr, 0, scale,
              (int)R, (int)S, dh,
              (long)C, 1, 1, (long)C, S, 1,
              Bn * NHh, NHh, (long)L * C, dh, (long)S * C, dh,
              (long)NHh * R * S, (long)R * S);
        softmax_rows<<<dim3((unsigned)(Bn * NHh * R)), dim3(256), 0, stream>>>(
            scb, (int)S);
        // O[b,l0+m,h*dh+d] = sum_s attn * V[b,s,h*dh+d]
        gemmL(stream, scb, Vb, Ob + l0 * C, nullptr, 0, 1.0f,
              (int)R, dh, (int)S,
              S, 1, (long)C, 1, (long)C, 1,
              Bn * NHh, NHh, (long)NHh * R * S, (long)R * S, (long)S * C, dh,
              (long)L * C, dh);
      }

      // out-proj, written transposed into cat[b, C*(at+1) + n, l]
      gemmL(stream, Ob, ow, catb + c0, ob, 1, 1.0f, (int)L, C, C,
            (long)C, 1, 1, (long)C, 1, L,
            Bn, 1, (long)L * C, 0, 0, 0, (long)3 * C * L, 0);
    }

    // squeeze: out[b,c,l] = sum_k sqw[c,k]*cat[b,k,l] + sqb[c]
    gemmL(stream, sqw, catb, outP + outOff, sqb, 2, 1.0f, C, (int)L, 3 * C,
          (long)3 * C, 1, L, 1, L, 1,
          Bn, 1, 0, 0, (long)3 * C * L, 0, (long)C * L, 0);
    outOff += (long)Bn * C * L;
  }
}
